// Attention_trad_37881611550825
// MI455X (gfx1250) — matrix operand order repaired and checked by reading, not compiled, not run on hardware
//
#include <hip/hip_runtime.h>
#include <hip/hip_bf16.h>

#define BB 32
#define SS 1024
#define EE 512
#define DD 512
#define UU 512

typedef __attribute__((ext_vector_type(16))) __bf16 v16bf;
typedef __attribute__((ext_vector_type(8)))  __bf16 v8bf;
typedef __attribute__((ext_vector_type(8)))  float  v8f;

__device__ __forceinline__ unsigned short f32_bf16_rne(float f) {
    union { float f; unsigned u; } v; v.f = f;
    unsigned u = v.u;
    u += 0x7FFFu + ((u >> 16) & 1u);   // round-to-nearest-even
    return (unsigned short)(u >> 16);
}

// ---------------------------------------------------------------------------
// Kernel 1: Ua [E][U] fp32  ->  UaT [U][E] bf16  (LDS 32x32 tile transpose)
// ---------------------------------------------------------------------------
__global__ void __launch_bounds__(256)
k_transpose_ua(const float* __restrict__ Ua, unsigned short* __restrict__ UaT) {
    __shared__ float tile[32][33];
    const int tu = blockIdx.x * 32;   // u tile origin
    const int te = blockIdx.y * 32;   // e tile origin
    const int tx = threadIdx.x;       // 0..31
    const int ty = threadIdx.y;       // 0..7
    #pragma unroll
    for (int i = 0; i < 32; i += 8)
        tile[ty + i][tx] = Ua[(size_t)(te + ty + i) * UU + (tu + tx)];
    __syncthreads();
    #pragma unroll
    for (int i = 0; i < 32; i += 8)
        UaT[(size_t)(tu + ty + i) * EE + (te + tx)] = f32_bf16_rne(tile[tx][ty + i]);
}

// ---------------------------------------------------------------------------
// Kernel 2: dec_proj[b][u] = sum_d hidden[b][d] * Wa[d][u]   (tiny, VALU)
// ---------------------------------------------------------------------------
__global__ void __launch_bounds__(256)
k_decproj(const float* __restrict__ hidden, const float* __restrict__ Wa,
          float* __restrict__ dp) {
    const int idx = blockIdx.x * 256 + threadIdx.x;   // 0 .. B*U-1
    const int b = idx / UU;
    const int u = idx % UU;
    float acc = 0.f;
    for (int d = 0; d < DD; ++d)
        acc = fmaf(hidden[b * DD + d], Wa[(size_t)d * UU + u], acc);
    dp[idx] = acc;
}

// ---------------------------------------------------------------------------
// Kernel 3: scores[b][s] = Va . tanh(enc[b,s,:] @ Ua + dec_proj[b,:])
// WMMA bf16 16x16x32, fused tanh/Va epilogue, per-wave 16-row M tile.
// ---------------------------------------------------------------------------
__global__ void __launch_bounds__(256)
k_scores(const float* __restrict__ enc, const unsigned short* __restrict__ UaT,
         const float* __restrict__ dp, const float* __restrict__ Va,
         float* __restrict__ scores) {
    const int lane  = threadIdx.x & 31;
    const int wave  = threadIdx.x >> 5;
    const int mBase = (blockIdx.x * 8 + wave) * 16;    // global row (b*S + s)
    const int b     = mBase / SS;                      // 16-row tile never crosses b
    const int hi16  = lane >> 4;                       // 0: lanes 0-15, 1: lanes 16-31
    const int l16   = lane & 15;

    // --- Load + convert A fragments once (16 K-steps, cached in VGPRs) -----
    // A 16x32 layout (contiguous, matches B): lanes 0-15 hold K[kb..kb+15],
    // lanes 16-31 hold K[kb+16..kb+31]. (kb = 32*k)
    union AF { v16bf v; v8bf h[2]; };
    AF a[16];
    const float* arow = enc + (size_t)(mBase + l16) * EE;
    #pragma unroll
    for (int k = 0; k < 16; ++k) {
        const int kb = k * 32 + hi16 * 16;
        #pragma unroll
        for (int j = 0; j < 8; ++j) {
            a[k].v[j]     = (__bf16)arow[kb + j];
            a[k].v[j + 8] = (__bf16)arow[kb + 8 + j];
        }
    }

    float sacc[8];
    #pragma unroll
    for (int i = 0; i < 8; ++i) sacc[i] = 0.f;

    // --- Loop N tiles of 16, K fully inside, fused epilogue ---------------
    for (int nT = 0; nT < 32; ++nT) {
        const int n = nT * 16 + l16;                   // this lane's N column
        // B fragment: lane holds 16 contiguous K values of column n
        const unsigned short* brow = UaT + (size_t)n * EE + hi16 * 16;
        v8f c = {};
        #pragma unroll
        for (int k = 0; k < 16; ++k) {
            v16bf bf = *reinterpret_cast<const v16bf*>(brow + k * 32);
            c = __builtin_amdgcn_wmma_f32_16x16x32_bf16(
                    false, a[k].v, false, bf, (short)0, c, false, false);
        }
        const float dpv = dp[b * UU + n];
        const float vav = Va[n];
        #pragma unroll
        for (int i = 0; i < 8; ++i) {                  // C: VGPR i -> M = i + 8*hi16
            float t = tanhf(c[i] + dpv);
            sacc[i] = fmaf(vav, t, sacc[i]);
        }
    }

    // --- Reduce over the 16 lanes of each half (each half = 8 M rows) -----
    #pragma unroll
    for (int i = 0; i < 8; ++i) {
        float v = sacc[i];
        v += __shfl_xor(v, 1);
        v += __shfl_xor(v, 2);
        v += __shfl_xor(v, 4);
        v += __shfl_xor(v, 8);
        if (l16 == 0) scores[mBase + hi16 * 8 + i] = v;
    }
}

// ---------------------------------------------------------------------------
// Kernel 4: softmax over S + context = alpha @ enc   (one block per batch b)
// ---------------------------------------------------------------------------
__global__ void __launch_bounds__(256)
k_softmax_ctx(const float* __restrict__ scores, const float* __restrict__ enc,
              float* __restrict__ out) {
    const int b = blockIdx.x;
    const int t = threadIdx.x;
    __shared__ float alpha[SS];
    __shared__ float red[256];

    float v[4];
    float m = -1e30f;
    #pragma unroll
    for (int i = 0; i < 4; ++i) {
        v[i] = scores[b * SS + t + i * 256];
        m = fmaxf(m, v[i]);
    }
    red[t] = m; __syncthreads();
    for (int off = 128; off > 0; off >>= 1) {
        if (t < off) red[t] = fmaxf(red[t], red[t + off]);
        __syncthreads();
    }
    m = red[0]; __syncthreads();

    float s = 0.f;
    #pragma unroll
    for (int i = 0; i < 4; ++i) {
        float e = __expf(v[i] - m);
        alpha[t + i * 256] = e;
        s += e;
    }
    red[t] = s; __syncthreads();
    for (int off = 128; off > 0; off >>= 1) {
        if (t < off) red[t] += red[t + off];
        __syncthreads();
    }
    const float inv = 1.0f / red[0];

    // context: each thread owns 2 E-columns, streams S coalesced
    for (int e0 = t; e0 < EE; e0 += 256) {
        float acc = 0.f;
        const float* ep = enc + (size_t)b * SS * EE + e0;
        for (int s2 = 0; s2 < SS; ++s2)
            acc = fmaf(alpha[s2], ep[(size_t)s2 * EE], acc);
        out[b * EE + e0] = acc * inv;
    }
}

// ---------------------------------------------------------------------------
extern "C" void kernel_launch(void* const* d_in, const int* in_sizes, int n_in,
                              void* d_out, int out_size, void* d_ws, size_t ws_size,
                              hipStream_t stream) {
    const float* enc    = (const float*)d_in[0];   // [B,S,E]
    const float* hidden = (const float*)d_in[1];   // [B,D]
    const float* Wa     = (const float*)d_in[2];   // [D,U]
    const float* Ua     = (const float*)d_in[3];   // [E,U]
    const float* Va     = (const float*)d_in[4];   // [U]
    float* out = (float*)d_out;                    // [B,E]

    char* ws = (char*)d_ws;
    unsigned short* UaT = (unsigned short*)ws;                               // 512 KB
    float* dp     = (float*)(ws + (size_t)EE * UU * sizeof(unsigned short)); // 64 KB
    float* scores = (float*)((char*)dp + (size_t)BB * UU * sizeof(float));   // 128 KB

    k_transpose_ua<<<dim3(UU / 32, EE / 32), dim3(32, 8), 0, stream>>>(Ua, UaT);
    k_decproj<<<(BB * UU) / 256, 256, 0, stream>>>(hidden, Wa, dp);
    k_scores<<<(BB * SS) / (16 * 8), 256, 0, stream>>>(enc, UaT, dp, Va, scores);
    k_softmax_ctx<<<BB, 256, 0, stream>>>(scores, enc, out);
}
